// LabelSmoothing_32993938768437
// MI455X (gfx1250) — matrix operand order repaired and checked by reading, not compiled, not run on hardware
//
#include <hip/hip_runtime.h>
#include <hip/hip_bf16.h>
#include <math.h>

typedef __attribute__((ext_vector_type(16))) _Float16 v16h;
typedef __attribute__((ext_vector_type(8)))  float    v8f;

namespace {
constexpr int kC  = 6;
constexpr int kH  = 512, kW = 512, kB = 8;
constexpr int kHW = kH * kW;                       // 262144
constexpr int kN  = kB * kHW;                      // 2097152
constexpr int kTilesX      = kW / 16;              // 32
constexpr int kTilesPerImg = kTilesX * (kH / 16);  // 1024
constexpr int kTiles       = kB * kTilesPerImg;    // 8192
constexpr int kWavesPerBlk = 8;
constexpr int kP1Blocks    = kTiles / kWavesPerBlk; // 1024
constexpr int kP2Blocks    = 1024;
constexpr int kP2Threads   = 256;

// workspace layout (bytes)
constexpr size_t OFF_COUNT = 0;                              // 1 x u32
constexpr size_t OFF_PA    = 64;                             // 1024 f32
constexpr size_t OFF_PU    = OFF_PA + kP2Blocks * 4;         // 1024 f32
constexpr size_t OFF_PV    = OFF_PU + kP2Blocks * 4;         // 1024 f32
constexpr size_t OFF_MASK  = 12416;                          // kN/8 = 262144 B of edge bits
} // namespace

__global__ void ls_init(unsigned* __restrict__ count) {
  if (threadIdx.x == 0) *count = 0u;
}

// ---------------------------------------------------------------------------
// Pass 1: 11x11 box sum via WMMA banded-ones matmuls (exact in f16).
// One wave32 per 16x16 output tile. edge_mask bits -> ws, count -> int atomic.
// ---------------------------------------------------------------------------
__global__ __launch_bounds__(256) void ls_edge_pass(const int* __restrict__ tgt,
                                                    unsigned* __restrict__ count,
                                                    unsigned short* __restrict__ maskhw) {
  __shared__ _Float16 Tb[kWavesPerBlk][32][32];   // labels, halo 6 each side of tile origin
  __shared__ _Float16 Hb[kWavesPerBlk][32][16];   // horizontal 11-sums
  __shared__ unsigned rm[kWavesPerBlk][16];       // per-row edge bitmasks

  const int tid  = threadIdx.x;
  const int wv   = tid >> 5;
  const int lane = tid & 31;
  const int tile  = blockIdx.x * kWavesPerBlk + wv;
  const int batch = tile / kTilesPerImg;
  const int t     = tile % kTilesPerImg;
  const int r0 = (t / kTilesX) * 16;
  const int c0 = (t % kTilesX) * 16;

  // Load 32x32 f16 label tile: image rows r0-6..r0+25, cols c0-6..c0+25 (zero pad).
  {
    const int  gc  = c0 - 6 + lane;
    const bool cok = (gc >= 0) & (gc < kW);
    const int  base = batch * kHW + gc;
    #pragma unroll 4
    for (int row = 0; row < 32; ++row) {
      const int gr = r0 - 6 + row;
      float v = 0.0f;
      if (cok && gr >= 0 && gr < kH) v = (float)tgt[base + gr * kW];
      Tb[wv][row][lane] = (_Float16)v;
    }
  }
  if (lane < 16) rm[wv][lane] = 0u;
  __syncthreads();

  const int mn  = lane & 15;               // M (A/D rows) or N (B/D cols)
  const int h8  = (lane < 16) ? 0 : 8;     // A-operand K offset per lane half
  const int k16 = (lane < 16) ? 0 : 16;    // unused; B operands also take K offset h8

  // Band matrix (B operand): B[k][n] = 1 iff 1 <= k-n <= 11.
  v16h bband;
  #pragma unroll
  for (int e = 0; e < 16; ++e) {
    const int d = (h8 + e + (e & 8)) - mn;
    bband[e] = (d >= 1 && d <= 11) ? (_Float16)1.0f : (_Float16)0.0f;
  }

  // Horizontal 11-sums for the two 16-row blocks: H = T(16x32) x Band(32x16).
  #pragma unroll
  for (int bb = 0; bb < 2; ++bb) {
    const _Float16* rp = &Tb[wv][bb * 16 + mn][0];
    v16h a;
    #pragma unroll
    for (int e = 0; e < 8; ++e) {      // A layout: lanes<16 K=e / K=16+e ; lanes>=16 +8
      a[e]     = rp[h8 + e];
      a[8 + e] = rp[16 + h8 + e];
    }
    v8f c = {};
    c = __builtin_amdgcn_wmma_f32_16x16x32_f16(false, a, false, bband,
                                               (short)0, c, false, false);
    #pragma unroll
    for (int g = 0; g < 8; ++g) {      // D layout: M = g + 8*(lane/16), N = lane%16
      const int m = g + 8 * (lane >> 4);
      Hb[wv][bb * 16 + m][mn] = (_Float16)c[g];
    }
  }
  __syncthreads();

  // Vertical 11-sums: V = Band(16x32) x H(32x16). Same band, as A operand.
  v16h av;
  #pragma unroll
  for (int e = 0; e < 8; ++e) {
    const int d0 = (h8 + e) - mn;
    const int d1 = (16 + h8 + e) - mn;
    av[e]     = (d0 >= 1 && d0 <= 11) ? (_Float16)1.0f : (_Float16)0.0f;
    av[8 + e] = (d1 >= 1 && d1 <= 11) ? (_Float16)1.0f : (_Float16)0.0f;
  }
  v16h bh;
  #pragma unroll
  for (int e = 0; e < 16; ++e) bh[e] = Hb[wv][h8 + e + (e & 8)][mn];
  v8f vsum = {};
  vsum = __builtin_amdgcn_wmma_f32_16x16x32_f16(false, av, false, bh,
                                                (short)0, vsum, false, false);

  // edge != 0  <=>  window sum != 121 * center label  (all values exact ints)
  #pragma unroll
  for (int g = 0; g < 8; ++g) {
    const int m = g + 8 * (lane >> 4);
    const float tc = (float)Tb[wv][6 + m][6 + mn];
    const unsigned bit = (vsum[g] != 121.0f * tc) ? 1u : 0u;
    atomicOr(&rm[wv][m], bit << mn);
  }
  __syncthreads();

  if (lane < 16) {
    maskhw[(batch * kHW + (r0 + lane) * kW + c0) >> 4] = (unsigned short)rm[wv][lane];
  }
  if (lane == 0) {
    unsigned tot = 0;
    #pragma unroll
    for (int r = 0; r < 16; ++r) tot += __popc(rm[wv][r]);
    atomicAdd(count, tot);   // integer atomic: order-independent, deterministic
  }
}

// ---------------------------------------------------------------------------
// Pass 2: stream x with b128 loads (4 pixels/thread/iter), fused log_softmax
// + masked accumulation. Deterministic fixed-order reductions.
// ---------------------------------------------------------------------------
__device__ __forceinline__ void ls_accum_pixel(float v0, float v1, float v2,
                                               float v3, float v4, float v5,
                                               int lab, unsigned ebit,
                                               float& aA, float& aU, float& aV) {
  const float mx = fmaxf(fmaxf(fmaxf(v0, v1), fmaxf(v2, v3)), fmaxf(v4, v5));
  const float se = __expf(v0 - mx) + __expf(v1 - mx) + __expf(v2 - mx) +
                   __expf(v3 - mx) + __expf(v4 - mx) + __expf(v5 - mx);
  const float lse = mx + __logf(se);
  const float sum = v0 + v1 + v2 + v3 + v4 + v5;
  float vl = v0;
  vl = (lab == 1) ? v1 : vl;
  vl = (lab == 2) ? v2 : vl;
  vl = (lab == 3) ? v3 : vl;
  vl = (lab == 4) ? v4 : vl;
  vl = (lab == 5) ? v5 : vl;
  const float lp = vl - lse;             // log_p at the label
  const float S  = sum - 6.0f * lse;     // sum_c log_p
  const float eb = (float)ebit;
  aA += lp;
  aU += eb * lp;
  aV += eb * S;
}

__global__ __launch_bounds__(kP2Threads) void ls_loss_pass(const float* __restrict__ x,
                                                           const int* __restrict__ tgt,
                                                           const unsigned* __restrict__ maskw,
                                                           float* __restrict__ pA,
                                                           float* __restrict__ pU,
                                                           float* __restrict__ pV) {
  float aA = 0.f, aU = 0.f, aV = 0.f;
  const int stride4 = kP2Blocks * kP2Threads * 4;   // 1048576 pixels/sweep
  for (int p = (blockIdx.x * kP2Threads + threadIdx.x) * 4; p < kN; p += stride4) {
    const int b   = p >> 18;
    const int idx = p & (kHW - 1);             // 16B aligned, no plane crossing
    const float* xb = x + (size_t)b * (kC * kHW) + idx;
    __builtin_prefetch(xb + stride4, 0, 1);    // global_prefetch at xb + 4*kHW (channel 4 of these pixels)

    const float4 f0 = *(const float4*)(xb + 0 * kHW);
    const float4 f1 = *(const float4*)(xb + 1 * kHW);
    const float4 f2 = *(const float4*)(xb + 2 * kHW);
    const float4 f3 = *(const float4*)(xb + 3 * kHW);
    const float4 f4 = *(const float4*)(xb + 4 * kHW);
    const float4 f5 = *(const float4*)(xb + 5 * kHW);
    const int4   lb = *(const int4*)(tgt + p);
    const unsigned mb = (maskw[p >> 5] >> (p & 31)) & 0xFu;

    ls_accum_pixel(f0.x, f1.x, f2.x, f3.x, f4.x, f5.x, lb.x, (mb >> 0) & 1u, aA, aU, aV);
    ls_accum_pixel(f0.y, f1.y, f2.y, f3.y, f4.y, f5.y, lb.y, (mb >> 1) & 1u, aA, aU, aV);
    ls_accum_pixel(f0.z, f1.z, f2.z, f3.z, f4.z, f5.z, lb.z, (mb >> 2) & 1u, aA, aU, aV);
    ls_accum_pixel(f0.w, f1.w, f2.w, f3.w, f4.w, f5.w, lb.w, (mb >> 3) & 1u, aA, aU, aV);
  }

  __shared__ float rA[kP2Threads], rU[kP2Threads], rV[kP2Threads];
  const int tid = threadIdx.x;
  rA[tid] = aA; rU[tid] = aU; rV[tid] = aV;
  __syncthreads();
  for (int s = kP2Threads / 2; s > 0; s >>= 1) {
    if (tid < s) { rA[tid] += rA[tid + s]; rU[tid] += rU[tid + s]; rV[tid] += rV[tid + s]; }
    __syncthreads();
  }
  if (tid == 0) { pA[blockIdx.x] = rA[0]; pU[blockIdx.x] = rU[0]; pV[blockIdx.x] = rV[0]; }
}

// ---------------------------------------------------------------------------
// Finalize: fixed-order reduction of 1024 partials, apply s, write -loss.
// ---------------------------------------------------------------------------
__global__ __launch_bounds__(256) void ls_finalize(const float* __restrict__ pA,
                                                   const float* __restrict__ pU,
                                                   const float* __restrict__ pV,
                                                   const unsigned* __restrict__ count,
                                                   float* __restrict__ out) {
  __shared__ float rA[256], rU[256], rV[256];
  const int tid = threadIdx.x;
  float a = 0.f, u = 0.f, v = 0.f;
  for (int i = tid; i < kP2Blocks; i += 256) { a += pA[i]; u += pU[i]; v += pV[i]; }
  rA[tid] = a; rU[tid] = u; rV[tid] = v;
  __syncthreads();
  for (int s = 128; s > 0; s >>= 1) {
    if (tid < s) { rA[tid] += rA[tid + s]; rU[tid] += rU[tid + s]; rV[tid] += rV[tid + s]; }
    __syncthreads();
  }
  if (tid == 0) {
    const float s = fminf((float)(*count) * (1.0f / (float)kN), 0.2f);
    // loss = mean( lp + edge*[(s/6 - 2s)*lp + s*S] )
    const float loss = (rA[0] + (s * (1.0f / 6.0f) - 2.0f * s) * rU[0] + s * rV[0])
                       * (1.0f / (float)kN);
    out[0] = -loss;
  }
}

extern "C" void kernel_launch(void* const* d_in, const int* in_sizes, int n_in,
                              void* d_out, int out_size, void* d_ws, size_t ws_size,
                              hipStream_t stream) {
  (void)in_sizes; (void)n_in; (void)out_size; (void)ws_size;
  const float* x   = (const float*)d_in[0];
  const int*   tgt = (const int*)d_in[1];

  char* ws = (char*)d_ws;
  unsigned*       count  = (unsigned*)(ws + OFF_COUNT);
  float*          pA     = (float*)(ws + OFF_PA);
  float*          pU     = (float*)(ws + OFF_PU);
  float*          pV     = (float*)(ws + OFF_PV);
  unsigned short* maskhw = (unsigned short*)(ws + OFF_MASK);
  const unsigned* maskw  = (const unsigned*)(ws + OFF_MASK);

  ls_init<<<1, 64, 0, stream>>>(count);
  ls_edge_pass<<<kP1Blocks, 256, 0, stream>>>(tgt, count, maskhw);
  ls_loss_pass<<<kP2Blocks, kP2Threads, 0, stream>>>(x, tgt, maskw, pA, pU, pV);
  ls_finalize<<<1, 256, 0, stream>>>(pA, pU, pV, count, (float*)d_out);
}
